// DiTBlock_84439057039862
// MI455X (gfx1250) — matrix operand order repaired and checked by reading, not compiled, not run on hardware
//
#include <hip/hip_runtime.h>
#include <hip/hip_bf16.h>
#include <math.h>
#include <stdint.h>

// ---------------------------------------------------------------------------
// DiT graph block for MI455X (gfx1250, wave32).
// GEMMs: bf16 WMMA 16x16x32 with f32 accumulation. B tiles staged in LDS via
// gfx1250 async-to-LDS copies (ASYNCcnt), double buffered; A streamed from
// global. Edge attention: wave-per-edge gather + shfl reductions + atomics.
// ---------------------------------------------------------------------------

typedef __attribute__((ext_vector_type(16))) __bf16 v16bf;
typedef __attribute__((ext_vector_type(8)))  __bf16 v8bf;
typedef __attribute__((ext_vector_type(8)))  float  v8f;
typedef int v4i_gcc __attribute__((vector_size(16)));   // matches builtin param

union Frag { v16bf v; v8bf h[2]; };

#define HID 256
#define NH  8
#define HD  32
#define MODLD 1536

#define AS1 __attribute__((address_space(1)))
#define AS3 __attribute__((address_space(3)))

#if defined(__has_builtin)
#if __has_builtin(__builtin_amdgcn_global_load_async_to_lds_b128) && \
    __has_builtin(__builtin_amdgcn_s_wait_asynccnt)
#define HAVE_ASYNC_LDS 1
#endif
#endif

// copy 16 bytes global -> LDS (async if available)
__device__ __forceinline__ void stage16(const void* g, void* l) {
#ifdef HAVE_ASYNC_LDS
  __builtin_amdgcn_global_load_async_to_lds_b128(
      (AS1 v4i_gcc*)(uintptr_t)g,
      (AS3 v4i_gcc*)(unsigned)(uintptr_t)l, 0, 0);
#else
  *(float4*)l = *(const float4*)g;
#endif
}

__device__ __forceinline__ void stage_wait() {
#ifdef HAVE_ASYNC_LDS
  __builtin_amdgcn_s_wait_asynccnt(0);
#endif
}

// ---------------- small utility kernels ----------------

__global__ void k_fill_f32(float* p, float v, long long n) {
  long long i = (long long)blockIdx.x * blockDim.x + threadIdx.x;
  if (i < n) p[i] = v;
}

__global__ void k_copy_f32(const float* a, float* b, int n) {
  int i = blockIdx.x * blockDim.x + threadIdx.x;
  if (i < n) b[i] = a[i];
}

// W is [K x N] row-major fp32 -> Wt is [N x K] row-major bf16
__global__ void k_transpose_bf16(const float* __restrict__ W,
                                 __bf16* __restrict__ Wt, int K, int N) {
  long long i = (long long)blockIdx.x * blockDim.x + threadIdx.x;
  if (i >= (long long)K * N) return;
  int n = (int)(i % N);
  int k = (int)(i / N);
  Wt[(long long)n * K + k] = (__bf16)W[i];
}

__global__ void k_silu_bf16(const float* __restrict__ c,
                            __bf16* __restrict__ out, long long n) {
  long long i = (long long)blockIdx.x * blockDim.x + threadIdx.x;
  if (i >= n) return;
  float v = c[i];
  out[i] = (__bf16)(v / (1.f + __expf(-v)));
}

__global__ void k_f32_to_bf16(const float* __restrict__ a,
                              __bf16* __restrict__ b, long long n) {
  long long i = (long long)blockIdx.x * blockDim.x + threadIdx.x;
  if (i < n) b[i] = (__bf16)a[i];
}

// LayerNorm (eps=1e-6, no affine) + adaLN modulate -> bf16. One block per row.
__global__ void k_ln_modulate(const float* __restrict__ x,
                              const float* __restrict__ mod,
                              int shift_off, int scale_off,
                              __bf16* __restrict__ out) {
  __shared__ float red[HID];
  int row = blockIdx.x;
  int col = threadIdx.x;
  float v = x[(long long)row * HID + col];
  red[col] = v;
  __syncthreads();
  for (int s = HID / 2; s > 0; s >>= 1) {
    if (col < s) red[col] += red[col + s];
    __syncthreads();
  }
  float mean = red[0] * (1.f / HID);
  __syncthreads();
  float d = v - mean;
  red[col] = d * d;
  __syncthreads();
  for (int s = HID / 2; s > 0; s >>= 1) {
    if (col < s) red[col] += red[col + s];
    __syncthreads();
  }
  float var = red[0] * (1.f / HID);
  float rs = rsqrtf(var + 1e-6f);
  float shift = mod[(long long)row * MODLD + shift_off + col];
  float scale = mod[(long long)row * MODLD + scale_off + col];
  float h = d * rs * (1.f + scale) + shift;
  out[(long long)row * HID + col] = (__bf16)h;
}

// ---------------- WMMA GEMM (bf16 in, f32 acc), fused epilogues ----------------
// A: [M x K] bf16 row-major.  Bt: [N x K] bf16 row-major (pre-transposed weight).
// Block: 256 threads = 8 waves; block tile 128(M) x 64(N); wave tile 16 x 64.
// B tile (64 x 32 bf16 = 4KB) staged in LDS, double buffered, async copies.
// LDS row stride 80B (20 banks) -> conflict-free b128 fragment reads.
// mode 0: out_f32 = acc + bias
// mode 1: out_bf16 = gelu_tanh(acc + bias)
// mode 2: out_f32 = resid + gate * (acc + bias), gate = mod[row][gate_off+col]

#define BROW 80                  // padded LDS bytes per B row (64 data + 16 pad)
#define BBUF (64 * BROW)         // one B buffer = 5120B

__device__ __forceinline__ v16bf load_afrag(const __bf16* __restrict__ A,
                                            long long row, int K, int kb, int half) {
  Frag f;
  const __bf16* p = A + row * (long long)K + kb + half * 8;
  f.h[0] = *(const v8bf*)(p);        // K = kb + half*8   .. +7
  f.h[1] = *(const v8bf*)(p + 16);   // K = kb+16+half*8  .. +7
  return f.v;
}

__device__ __forceinline__ v16bf load_bfrag_lds(const char* base, int nloc, int half) {
  Frag f;
  const char* p = base + nloc * BROW + half * 16;
  f.h[0] = *(const v8bf*)(p);        // K = half*8      .. +7
  f.h[1] = *(const v8bf*)(p + 32);   // K = 16 + half*8 .. +7
  return f.v;
}

__global__ __launch_bounds__(256)
void k_gemm_wmma(const __bf16* __restrict__ A, const __bf16* __restrict__ Bt,
                 const float* __restrict__ bias, int M, int N, int K, int mode,
                 float* __restrict__ outf, __bf16* __restrict__ outb,
                 const float* __restrict__ resid, const float* __restrict__ mod,
                 int gate_off) {
  __shared__ __align__(16) char ldsB[2 * BBUF];

  const int wave = threadIdx.x >> 5;
  const int lane = threadIdx.x & 31;
  const int half = lane >> 4;
  const int lm   = lane & 15;

  const int m0 = blockIdx.x * 128 + wave * 16;
  const int n0 = blockIdx.y * 64;

  // clamp A row so OOB tiles never fault; results masked at the store.
  long long arow = m0 + lm;
  if (arow > M - 1) arow = M - 1;
  const __bf16* aptr = A + arow * (long long)K;

  // staging: 256 threads x 16B == one full 64x32 bf16 B tile per k-step.
  const int srow   = threadIdx.x >> 2;            // 0..63: B-tile row (n0+srow)
  const int schunk = (threadIdx.x & 3) * 16;      // byte offset within 64B row
  const char* gstage = (const char*)(Bt + (long long)(n0 + srow) * K) + schunk;
  char*       lstage = ldsB + srow * BROW + schunk;

  v8f acc0 = {}, acc1 = {}, acc2 = {}, acc3 = {};

  // prime buffer 0 with kb=0
  stage16(gstage, lstage);
  stage_wait();
  __syncthreads();

  int buf = 0;
  for (int kb = 0; kb < K; kb += 32) {
    if (kb + 32 < K) {
      // overlap next B-tile copy with this step's compute
      stage16(gstage + (long long)(kb + 32) * 2, lstage + (buf ^ 1) * BBUF);
      __builtin_prefetch(aptr + kb + 32, 0, 3);
    }
    v16bf a = load_afrag(A, arow, K, kb, half);
    const char* bbase = ldsB + buf * BBUF;
    v16bf b0 = load_bfrag_lds(bbase, 0 * 16 + lm, half);
    v16bf b1 = load_bfrag_lds(bbase, 1 * 16 + lm, half);
    v16bf b2 = load_bfrag_lds(bbase, 2 * 16 + lm, half);
    v16bf b3 = load_bfrag_lds(bbase, 3 * 16 + lm, half);
    acc0 = __builtin_amdgcn_wmma_f32_16x16x32_bf16(false, a, false, b0, (short)0, acc0, false, false);
    acc1 = __builtin_amdgcn_wmma_f32_16x16x32_bf16(false, a, false, b1, (short)0, acc1, false, false);
    acc2 = __builtin_amdgcn_wmma_f32_16x16x32_bf16(false, a, false, b2, (short)0, acc2, false, false);
    acc3 = __builtin_amdgcn_wmma_f32_16x16x32_bf16(false, a, false, b3, (short)0, acc3, false, false);
    stage_wait();
    __syncthreads();
    buf ^= 1;
  }

  v8f accs[4] = {acc0, acc1, acc2, acc3};
#pragma unroll
  for (int j = 0; j < 4; j++) {
    const int col = n0 + j * 16 + lm;
    const float bv = bias ? bias[col] : 0.f;
#pragma unroll
    for (int r = 0; r < 8; r++) {
      const int row = m0 + r + half * 8;   // D layout: VGPR r -> M = r + 8*(lane>=16)
      if (row < M) {
        float v = accs[j][r] + bv;
        long long oidx = (long long)row * N + col;
        if (mode == 0) {
          outf[oidx] = v;
        } else if (mode == 1) {
          float t = 0.7978845608028654f * (v + 0.044715f * v * v * v);
          float g = 0.5f * v * (1.f + tanhf(t));
          outb[oidx] = (__bf16)g;
        } else {
          float gate = mod[(long long)row * MODLD + gate_off + col];
          outf[oidx] = resid[(long long)row * N + col] + gate * v;
        }
      }
    }
  }
}

// ---------------- edge attention kernels ----------------

__device__ __forceinline__ void atomicMaxFloat(float* addr, float val) {
  if (val >= 0.f)
    atomicMax((int*)addr, __float_as_int(val));
  else
    atomicMin((unsigned int*)addr, __float_as_uint(val));
}

// One wave per edge: sim[e,h] = (q[dst,h,:] . k[src,h,:]) / sqrt(HD); atomic seg-max.
__global__ void k_edge_sim(const int* __restrict__ ei, const float* __restrict__ qkv,
                           float* __restrict__ sim, float* __restrict__ smax, int E) {
  int e = blockIdx.x * (blockDim.x >> 5) + (threadIdx.x >> 5);
  if (e >= E) return;
  int lane = threadIdx.x & 31;
  int src = ei[e];
  int dst = ei[E + e];
  const float* qd = qkv + (long long)dst * 768 + lane * 8;          // q at +0
  const float* ks = qkv + (long long)src * 768 + 256 + lane * 8;    // k at +256
  float4 q0 = *(const float4*)(qd);
  float4 q1 = *(const float4*)(qd + 4);
  float4 k0 = *(const float4*)(ks);
  float4 k1 = *(const float4*)(ks + 4);
  float p = q0.x * k0.x + q0.y * k0.y + q0.z * k0.z + q0.w * k0.w
          + q1.x * k1.x + q1.y * k1.y + q1.z * k1.z + q1.w * k1.w;
  p += __shfl_xor(p, 1, 32);
  p += __shfl_xor(p, 2, 32);          // lanes 4h..4h+3 now hold head-h dot
  if ((lane & 3) == 0) {
    int h = lane >> 2;
    float s = p * 0.17677669529663687f;   // HD^-0.5
    sim[(long long)e * NH + h] = s;
    atomicMaxFloat(&smax[(long long)dst * NH + h], s);
  }
}

// Thread per (edge, head): exp(sim - smax[dst]); atomic seg-sum for denom.
__global__ void k_edge_exp(const int* __restrict__ ei, float* __restrict__ sim,
                           const float* __restrict__ smax, float* __restrict__ denom,
                           int E) {
  long long idx = (long long)blockIdx.x * blockDim.x + threadIdx.x;
  if (idx >= (long long)E * NH) return;
  int e = (int)(idx >> 3);
  int h = (int)(idx & 7);
  int dst = ei[E + e];
  float w = __expf(sim[idx] - smax[(long long)dst * NH + h]);
  sim[idx] = w;
  atomicAdd(&denom[(long long)dst * NH + h], w);
}

// One wave per edge: agg[dst] += v[src] * attn.
__global__ void k_edge_agg(const int* __restrict__ ei, const float* __restrict__ qkv,
                           const float* __restrict__ sim, const float* __restrict__ denom,
                           float* __restrict__ agg, int E) {
  int e = blockIdx.x * (blockDim.x >> 5) + (threadIdx.x >> 5);
  if (e >= E) return;
  int lane = threadIdx.x & 31;
  int src = ei[e];
  int dst = ei[E + e];
  int h = lane >> 2;
  float attn = sim[(long long)e * NH + h] / denom[(long long)dst * NH + h];
  const float* vs = qkv + (long long)src * 768 + 512 + lane * 8;    // v at +512
  float* ag = agg + (long long)dst * HID + lane * 8;
  float4 v0 = *(const float4*)(vs);
  float4 v1 = *(const float4*)(vs + 4);
  atomicAdd(ag + 0, v0.x * attn);
  atomicAdd(ag + 1, v0.y * attn);
  atomicAdd(ag + 2, v0.z * attn);
  atomicAdd(ag + 3, v0.w * attn);
  atomicAdd(ag + 4, v1.x * attn);
  atomicAdd(ag + 5, v1.y * attn);
  atomicAdd(ag + 6, v1.z * attn);
  atomicAdd(ag + 7, v1.w * attn);
}

// ---------------------------------------------------------------------------

static inline unsigned cdiv(long long a, long long b) { return (unsigned)((a + b - 1) / b); }

extern "C" void kernel_launch(void* const* d_in, const int* in_sizes, int n_in,
                              void* d_out, int out_size, void* d_ws, size_t ws_size,
                              hipStream_t stream) {
  const float* x  = (const float*)d_in[0];
  const float* c  = (const float*)d_in[1];
  const int*   ei = (const int*)d_in[2];
  const float* Wq = (const float*)d_in[3];  const float* bq = (const float*)d_in[4];
  const float* Wk = (const float*)d_in[5];  const float* bk = (const float*)d_in[6];
  const float* Wv = (const float*)d_in[7];  const float* bv = (const float*)d_in[8];
  const float* Wo = (const float*)d_in[9];  const float* bo = (const float*)d_in[10];
  const float* W1 = (const float*)d_in[11]; const float* b1 = (const float*)d_in[12];
  const float* W2 = (const float*)d_in[13]; const float* b2 = (const float*)d_in[14];
  const float* Wc = (const float*)d_in[15]; const float* bc = (const float*)d_in[16];

  const int M = in_sizes[0] / HID;     // 50000
  const int E = in_sizes[2] / 2;       // 800000

  char* ws = (char*)d_ws;
  size_t off = 0;
  auto alloc = [&](size_t bytes) {
    size_t o = off;
    off = (off + bytes + 255) & ~(size_t)255;
    return o;
  };
  const size_t oWcT   = alloc((size_t)1536 * HID * 2);
  const size_t oWqkvT = alloc((size_t)768 * HID * 2);
  const size_t oWoT   = alloc((size_t)HID * HID * 2);
  const size_t oW1T   = alloc((size_t)1024 * HID * 2);
  const size_t oW2T   = alloc((size_t)HID * 1024 * 2);
  const size_t oBqkv  = alloc((size_t)768 * 4);
  const size_t oScB   = alloc((size_t)M * HID * 2);
  const size_t oMod   = alloc((size_t)M * MODLD * 4);
  const size_t oHB    = alloc((size_t)M * HID * 2);
  const size_t oQKV   = alloc((size_t)M * 768 * 4);
  const size_t oSim   = alloc((size_t)E * NH * 4);
  const size_t oSmax  = alloc((size_t)M * NH * 4);
  const size_t oDen   = alloc((size_t)M * NH * 4);
  const size_t oAgg   = alloc((size_t)M * HID * 4);
  const size_t oAggB  = alloc((size_t)M * HID * 2);
  const size_t oXmid  = alloc((size_t)M * HID * 4);
  const size_t oHmlpB = alloc((size_t)M * HID * 2);
  const size_t oH2B   = alloc((size_t)M * 1024 * 2);
  (void)ws_size; (void)n_in; (void)out_size;

  __bf16* WcT   = (__bf16*)(ws + oWcT);
  __bf16* WqkvT = (__bf16*)(ws + oWqkvT);
  __bf16* WoT   = (__bf16*)(ws + oWoT);
  __bf16* W1T   = (__bf16*)(ws + oW1T);
  __bf16* W2T   = (__bf16*)(ws + oW2T);
  float*  bqkv  = (float*)(ws + oBqkv);
  __bf16* scB   = (__bf16*)(ws + oScB);
  float*  modb  = (float*)(ws + oMod);
  __bf16* hB    = (__bf16*)(ws + oHB);
  float*  qkv   = (float*)(ws + oQKV);
  float*  sim   = (float*)(ws + oSim);
  float*  smax  = (float*)(ws + oSmax);
  float*  den   = (float*)(ws + oDen);
  float*  agg   = (float*)(ws + oAgg);
  __bf16* aggB  = (__bf16*)(ws + oAggB);
  float*  xmid  = (float*)(ws + oXmid);
  __bf16* hmlpB = (__bf16*)(ws + oHmlpB);
  __bf16* h2B   = (__bf16*)(ws + oH2B);

  const dim3 blk(256);

  // 1) weights -> transposed bf16 (stay resident in 192MB L2 afterwards)
  k_transpose_bf16<<<cdiv((long long)HID * 1536, 256), blk, 0, stream>>>(Wc, WcT, HID, 1536);
  k_transpose_bf16<<<cdiv((long long)HID * HID, 256), blk, 0, stream>>>(Wq, WqkvT + 0 * 256 * HID, HID, HID);
  k_transpose_bf16<<<cdiv((long long)HID * HID, 256), blk, 0, stream>>>(Wk, WqkvT + 1 * 256 * HID, HID, HID);
  k_transpose_bf16<<<cdiv((long long)HID * HID, 256), blk, 0, stream>>>(Wv, WqkvT + 2 * 256 * HID, HID, HID);
  k_transpose_bf16<<<cdiv((long long)HID * HID, 256), blk, 0, stream>>>(Wo, WoT, HID, HID);
  k_transpose_bf16<<<cdiv((long long)HID * 1024, 256), blk, 0, stream>>>(W1, W1T, HID, 1024);
  k_transpose_bf16<<<cdiv((long long)1024 * HID, 256), blk, 0, stream>>>(W2, W2T, 1024, HID);
  k_copy_f32<<<1, 256, 0, stream>>>(bq, bqkv + 0, 256);
  k_copy_f32<<<1, 256, 0, stream>>>(bk, bqkv + 256, 256);
  k_copy_f32<<<1, 256, 0, stream>>>(bv, bqkv + 512, 256);

  // 2) mod = silu(c) @ Wc + bc
  k_silu_bf16<<<cdiv((long long)M * HID, 256), blk, 0, stream>>>(c, scB, (long long)M * HID);
  {
    dim3 g(cdiv(M, 128), 1536 / 64);
    k_gemm_wmma<<<g, blk, 0, stream>>>(scB, WcT, bc, M, 1536, HID, 0,
                                       modb, nullptr, nullptr, nullptr, 0);
  }

  // 3) h = modulate(LN(x), shift_msa, scale_msa);  qkv = h @ [Wq|Wk|Wv] + b
  k_ln_modulate<<<M, HID, 0, stream>>>(x, modb, 0, 256, hB);
  {
    dim3 g(cdiv(M, 128), 768 / 64);
    k_gemm_wmma<<<g, blk, 0, stream>>>(hB, WqkvT, bqkv, M, 768, HID, 0,
                                       qkv, nullptr, nullptr, nullptr, 0);
  }

  // 4) edge attention with segment softmax
  k_fill_f32<<<cdiv((long long)M * NH, 256), blk, 0, stream>>>(smax, -INFINITY, (long long)M * NH);
  k_fill_f32<<<cdiv((long long)M * NH, 256), blk, 0, stream>>>(den, 0.f, (long long)M * NH);
  k_fill_f32<<<cdiv((long long)M * HID, 256), blk, 0, stream>>>(agg, 0.f, (long long)M * HID);
  k_edge_sim<<<cdiv(E, 8), blk, 0, stream>>>(ei, qkv, sim, smax, E);
  k_edge_exp<<<cdiv((long long)E * NH, 256), blk, 0, stream>>>(ei, sim, smax, den, E);
  k_edge_agg<<<cdiv(E, 8), blk, 0, stream>>>(ei, qkv, sim, den, agg, E);

  // 5) xmid = x + gate_msa * (agg @ Wo + bo)
  k_f32_to_bf16<<<cdiv((long long)M * HID, 256), blk, 0, stream>>>(agg, aggB, (long long)M * HID);
  {
    dim3 g(cdiv(M, 128), HID / 64);
    k_gemm_wmma<<<g, blk, 0, stream>>>(aggB, WoT, bo, M, HID, HID, 2,
                                       xmid, nullptr, x, modb, 2 * 256);
  }

  // 6) MLP: out = xmid + gate_mlp * (gelu(mod(LN(xmid)) @ W1 + b1) @ W2 + b2)
  k_ln_modulate<<<M, HID, 0, stream>>>(xmid, modb, 3 * 256, 4 * 256, hmlpB);
  {
    dim3 g(cdiv(M, 128), 1024 / 64);
    k_gemm_wmma<<<g, blk, 0, stream>>>(hmlpB, W1T, b1, M, 1024, HID, 1,
                                       nullptr, h2B, nullptr, nullptr, 0);
  }
  {
    dim3 g(cdiv(M, 128), HID / 64);
    k_gemm_wmma<<<g, blk, 0, stream>>>(h2B, W2T, b2, M, HID, 1024, 2,
                                       (float*)d_out, nullptr, xmid, modb, 5 * 256);
  }
}
